// MixtralSparseMoe_13125420057239
// MI455X (gfx1250) — matrix operand order repaired and checked by reading, not compiled, not run on hardware
//
#include <hip/hip_runtime.h>

// ---------------------------------------------------------------------------
// MixtralSparseMoe for MI455X (gfx1250, wave32, WMMA bf16 + TDM gather)
//   T=2048 tokens, D=1024, F=4096, E=8 experts, top-2 routing.
//   d_out = [out (T*D f32)] ++ [router_logits (T*E f32)]
// ---------------------------------------------------------------------------

#define TT 2048
#define DD 1024
#define FF 4096
#define EE 8
#define BM 32     // tokens per tile (2 WMMA row-tiles)
#define FB 128    // F-chunk
#define KB 32     // WMMA K step (bf16)

typedef __bf16 bf16;
typedef __attribute__((ext_vector_type(16))) __bf16 v16bf;
typedef __attribute__((ext_vector_type(8)))  __bf16 v8bf;
typedef __attribute__((ext_vector_type(8)))  float  v8f;
typedef __attribute__((ext_vector_type(4)))  unsigned int u32x4;
typedef __attribute__((ext_vector_type(8)))  int i32x8;
typedef __attribute__((ext_vector_type(4)))  int i32x4;

static __device__ __forceinline__ bf16 f2bf(float f) { return (bf16)f; }

// ---------------------------------------------------------------------------
// init: zero the main output, combine weights, expert counters
// ---------------------------------------------------------------------------
__global__ void moe_init_kernel(float* __restrict__ out,
                                float* __restrict__ combine,
                                int* __restrict__ cnt) {
  size_t i = (size_t)blockIdx.x * blockDim.x + threadIdx.x;
  if (i < (size_t)TT * DD) out[i] = 0.0f;
  if (i < (size_t)TT * EE) combine[i] = 0.0f;
  if (i < EE) cnt[i] = 0;
}

// ---------------------------------------------------------------------------
// cast activations fp32 -> bf16 (streamed once; 4 MB workspace)
// ---------------------------------------------------------------------------
__global__ void moe_cast_kernel(const float* __restrict__ x,
                                bf16* __restrict__ xh) {
  size_t i = (size_t)blockIdx.x * blockDim.x + threadIdx.x;
  if (i < (size_t)TT * DD) xh[i] = f2bf(x[i]);
}

// ---------------------------------------------------------------------------
// router: one wave32 per token. logits -> softmax -> top2 -> renorm.
// Builds per-expert token lists with atomics.
// ---------------------------------------------------------------------------
__global__ void moe_router_kernel(const float* __restrict__ x,
                                  const float* __restrict__ gw,
                                  float* __restrict__ logits_out,
                                  float* __restrict__ combine,
                                  int* __restrict__ cnt,
                                  int* __restrict__ tokens) {
  int wave = threadIdx.x >> 5;
  int lane = threadIdx.x & 31;
  int t = blockIdx.x * 8 + wave;
  if (t >= TT) return;

  const float4* xr = (const float4*)(x + (size_t)t * DD);  // 256 float4
  float lg[EE];
#pragma unroll
  for (int e = 0; e < EE; ++e) {
    const float4* gr = (const float4*)(gw + (size_t)e * DD);
    float s = 0.0f;
    for (int d = lane; d < DD / 4; d += 32) {
      float4 a = xr[d], b = gr[d];
      s += a.x * b.x + a.y * b.y + a.z * b.z + a.w * b.w;
    }
#pragma unroll
    for (int off = 16; off > 0; off >>= 1) s += __shfl_xor(s, off, 32);
    lg[e] = s;
  }

  if (lane == 0) {
    float m = lg[0];
#pragma unroll
    for (int e = 1; e < EE; ++e) m = fmaxf(m, lg[e]);
    float p[EE];
#pragma unroll
    for (int e = 0; e < EE; ++e) p[e] = __expf(lg[e] - m);
    // top-2 (softmax denominator cancels in the renormalization)
    int e1 = 0;
#pragma unroll
    for (int e = 1; e < EE; ++e) if (p[e] > p[e1]) e1 = e;
    int e2 = (e1 == 0) ? 1 : 0;
#pragma unroll
    for (int e = 0; e < EE; ++e) if (e != e1 && p[e] > p[e2]) e2 = e;
    float denom = p[e1] + p[e2];

#pragma unroll
    for (int e = 0; e < EE; ++e) logits_out[(size_t)t * EE + e] = lg[e];
    combine[(size_t)t * EE + e1] = p[e1] / denom;
    combine[(size_t)t * EE + e2] = p[e2] / denom;

    int p1 = atomicAdd(&cnt[e1], 1);
    tokens[(size_t)e1 * TT + p1] = t;
    int p2 = atomicAdd(&cnt[e2], 1);
    tokens[(size_t)e2 * TT + p2] = t;
  }
}

// ---------------------------------------------------------------------------
// Expert SwiGLU FFN over gathered tokens.
// Block = (expert e, 32-token tile). 256 threads = 8 waves.
// Wave w owns: phase1 f-cols [fc + 16w, +16) x 2 row-tiles;
//              phase2 out-cols [128w, +128) x 2 row-tiles.
//
// Token-row gather: two TDM gather descriptors (16x 16-bit indices each)
// pull 32 scattered 2KB bf16 rows from xh into LDS (TENSORcnt tracked).
// B-operand slabs staged fragment-major [ntile][lane][16 bf16]: each lane's
// fragment is one contiguous 32B LDS read, reused for both row-tiles.
// ---------------------------------------------------------------------------
__global__ void __launch_bounds__(256)
moe_expert_kernel(const float* __restrict__ w1,
                  const float* __restrict__ w2,
                  const float* __restrict__ w3,
                  const bf16* __restrict__ xh,
                  const float* __restrict__ combine,
                  const int* __restrict__ cnt,
                  const int* __restrict__ tokens,
                  float* __restrict__ out) {
  __shared__ __align__(32) bf16 xs[BM][DD];        // gathered rows   64 KB
  __shared__ __align__(32) bf16 ws1f[8][32][16];   // W1 frag-major    8 KB
  __shared__ __align__(32) bf16 ws3f[8][32][16];   // W3 frag-major    8 KB
  __shared__ __align__(32) bf16 asb[BM][FB];       // silu(h)*g tile   8 KB
  __shared__ __align__(32) bf16 w2sf[64][32][16];  // W2 frag-major   64 KB
  __shared__ int   trow[BM];
  __shared__ float crow[BM];

  const int e    = blockIdx.x / (TT / BM);
  const int tile = blockIdx.x % (TT / BM);
  const int n_e  = cnt[e];
  if (tile * BM >= n_e) return;

  const int tid = threadIdx.x;
  if (tid < BM) {
    int r = tile * BM + tid;
    int t = (r < n_e) ? tokens[(size_t)e * TT + r] : -1;
    trow[tid] = t;
    crow[tid] = (t >= 0) ? combine[(size_t)t * EE + e] : 0.0f;
  }
  __syncthreads();

#if __has_builtin(__builtin_amdgcn_tensor_load_to_lds) && \
    __has_builtin(__builtin_amdgcn_s_wait_tensorcnt)
  // ---- TDM gather: 2 x 16 scattered bf16 rows (2KB each) -> xs ----------
  if (tid < 32) {
    unsigned lds_addr = (unsigned)(size_t)(void*)&xs[0][0];
    unsigned long long ga = (unsigned long long)(size_t)xh;
#pragma unroll
    for (int half = 0; half < 2; ++half) {
      u32x4 g0;
      g0[0] = 0x80000001u;  // gather_mode=1, 16-bit indices, count=1
      g0[1] = (unsigned)__builtin_amdgcn_readfirstlane(
                  (int)(lds_addr + half * 16 * DD * 2));
      g0[2] = (unsigned)(ga & 0xffffffffu);
      g0[3] = ((unsigned)((ga >> 32) & 0x01ffffffu)) | 0x80000000u;  // type=2

      i32x8 g1;
      g1[0] = 0x00010000;          // data_size=1 (2 bytes), no multicast/pad
      g1[1] = (int)(1024u << 16);  // tensor_dim0[15:0] = D
      g1[2] = (int)(2048u << 16);  // tensor_dim0 hi=0 | tensor_dim1[15:0] = T
      g1[3] = (int)(1024u << 16);  // tensor_dim1 hi=0 | tile_dim0 = D
      g1[4] = 16;                  // tile_dim1 = #row indices
      g1[5] = 1024;                // tensor_dim0_stride = D
      g1[6] = 0;
      g1[7] = 0;

      i32x4 g2, g3;
#pragma unroll
      for (int j = 0; j < 4; ++j) {
        int lo = trow[half * 16 + 2 * j];      lo = (lo < 0) ? 0 : lo;
        int hi = trow[half * 16 + 2 * j + 1];  hi = (hi < 0) ? 0 : hi;
        g2[j] = __builtin_amdgcn_readfirstlane(lo | (hi << 16));
        int lo2 = trow[half * 16 + 8 + 2 * j];     lo2 = (lo2 < 0) ? 0 : lo2;
        int hi2 = trow[half * 16 + 8 + 2 * j + 1]; hi2 = (hi2 < 0) ? 0 : hi2;
        g3[j] = __builtin_amdgcn_readfirstlane(lo2 | (hi2 << 16));
      }
      i32x8 gz = {0, 0, 0, 0, 0, 0, 0, 0};  // unused trailing group
      __builtin_amdgcn_tensor_load_to_lds(g0, g1, g2, g3, gz, 0);
    }
    __builtin_amdgcn_s_wait_tensorcnt(0);
  }
#else
  // ---- fallback: vector-copy gather -------------------------------------
#pragma unroll 1
  for (int idx = tid; idx < BM * (DD / 8); idx += 256) {
    int m = idx >> 7;
    int c = idx & 127;
    int t = trow[m];
    uint4 v = make_uint4(0u, 0u, 0u, 0u);
    if (t >= 0) v = *(const uint4*)(xh + (size_t)t * DD + c * 8);
    *(uint4*)&xs[m][c * 8] = v;
  }
#endif
  __syncthreads();

  const int wave = tid >> 5;
  const int lane = tid & 31;
  const int lm   = lane & 15;   // M (A) or N (B, C/D) index
  const int lh   = lane >> 4;   // lane half -> K half

  v8f oacc0[8], oacc1[8];       // 2 row-tiles x 8 col-tiles
#pragma unroll
  for (int tn = 0; tn < 8; ++tn) {
    v8f z = {0.f, 0.f, 0.f, 0.f, 0.f, 0.f, 0.f, 0.f};
    oacc0[tn] = z;
    oacc1[tn] = z;
  }

#pragma unroll 1
  for (int fc = 0; fc < FF; fc += FB) {
    // ------- phase 1: h = x@W1, g = x@W3 for f-cols [fc, fc+FB) ----------
    v8f h0 = {0.f, 0.f, 0.f, 0.f, 0.f, 0.f, 0.f, 0.f};
    v8f h1 = h0, g0a = h0, g1a = h0;

#pragma unroll 1
    for (int k0 = 0; k0 < DD; k0 += KB) {
      __syncthreads();  // previous slab fully consumed
      // stage [KB][FB] slabs of W1/W3, fragment-major, float4 global reads
#pragma unroll
      for (int r = 0; r < 4; ++r) {
        int idx = tid + r * 256;        // float4 index within slab
        int kk = idx >> 5;              // row (128 floats = 32 float4/row)
        int fb = (idx & 31) * 4;        // col base
        size_t g = ((size_t)e * DD + (size_t)(k0 + kk)) * FF + (fc + fb);
        float4 a = *(const float4*)(w1 + g);
        float4 b = *(const float4*)(w3 + g);
        const float* pa = (const float*)&a;
        const float* pb = (const float*)&b;
#pragma unroll
        for (int u = 0; u < 4; ++u) {
          int ff = fb + u;
          int j  = ff >> 4;
          int ln = (ff & 15) | ((kk >> 4) << 4);
          int ii = kk & 15;
          ws1f[j][ln][ii] = f2bf(pa[u]);
          ws3f[j][ln][ii] = f2bf(pb[u]);
        }
      }
      // prefetch next k-slab while this one computes
      if (k0 + KB < DD) {
        size_t gn = ((size_t)e * DD + (size_t)(k0 + KB) + (tid >> 4)) * FF
                    + fc + (tid & 15) * 8;
        __builtin_prefetch(w1 + gn, 0, 1);
        __builtin_prefetch(w3 + gn, 0, 1);
      }
      __syncthreads();

      // A fragments (rows 0-15 and 16-31): two contiguous 16B runs each
      v8bf a0l = *(const v8bf*)&xs[lm][k0 + lh * 16];
      v8bf a0h = *(const v8bf*)&xs[lm][k0 + lh * 16 + 8];
      v16bf a0 = __builtin_shufflevector(a0l, a0h,
                                         0, 1, 2, 3, 4, 5, 6, 7,
                                         8, 9, 10, 11, 12, 13, 14, 15);
      v8bf a1l = *(const v8bf*)&xs[16 + lm][k0 + lh * 16];
      v8bf a1h = *(const v8bf*)&xs[16 + lm][k0 + lh * 16 + 8];
      v16bf a1 = __builtin_shufflevector(a1l, a1h,
                                         0, 1, 2, 3, 4, 5, 6, 7,
                                         8, 9, 10, 11, 12, 13, 14, 15);
      // B fragments: one contiguous 32B run per lane, reused by both tiles
      v16bf b1 = *(const v16bf*)&ws1f[wave][lane][0];
      v16bf b3 = *(const v16bf*)&ws3f[wave][lane][0];

      h0  = __builtin_amdgcn_wmma_f32_16x16x32_bf16(false, a0, false, b1,
                                                    (short)0, h0, false, false);
      h1  = __builtin_amdgcn_wmma_f32_16x16x32_bf16(false, a1, false, b1,
                                                    (short)0, h1, false, false);
      g0a = __builtin_amdgcn_wmma_f32_16x16x32_bf16(false, a0, false, b3,
                                                    (short)0, g0a, false, false);
      g1a = __builtin_amdgcn_wmma_f32_16x16x32_bf16(false, a1, false, b3,
                                                    (short)0, g1a, false, false);
    }

    // ------- SwiGLU: a = silu(h) * g  ->  bf16 tile in LDS ---------------
    __syncthreads();
#pragma unroll
    for (int v = 0; v < 8; ++v) {
      int M = v + 8 * lh;                 // f32 C/D layout
      float ha = h0[v], ga = g0a[v];
      float sa = ha * __builtin_amdgcn_rcpf(1.0f + __expf(-ha));
      asb[M][wave * 16 + lm] = f2bf(sa * ga);
      float hb = h1[v], gb = g1a[v];
      float sb = hb * __builtin_amdgcn_rcpf(1.0f + __expf(-hb));
      asb[16 + M][wave * 16 + lm] = f2bf(sb * gb);
    }
    __syncthreads();

    // ------- phase 2: out += a_chunk @ W2[fc:fc+FB, :] --------------------
#pragma unroll 1
    for (int kk = 0; kk < FB; kk += KB) {
      // stage [KB][DD] slab of W2, fragment-major, float4 global reads
#pragma unroll 1
      for (int r8 = 0; r8 < 32; ++r8) {
        int idx = tid + r8 * 256;       // float4 index within slab
        int rr = idx >> 8;              // row (1024 floats = 256 float4/row)
        int cb = (idx & 255) * 4;       // col base
        size_t g = ((size_t)e * FF + (size_t)(fc + kk + rr)) * DD + cb;
        float4 a = *(const float4*)(w2 + g);
        const float* pa = (const float*)&a;
#pragma unroll
        for (int u = 0; u < 4; ++u) {
          int cc = cb + u;
          int j  = cc >> 4;
          int ln = (cc & 15) | ((rr >> 4) << 4);
          int ii = rr & 15;
          w2sf[j][ln][ii] = f2bf(pa[u]);
        }
      }
      if (kk + KB < FB) {
        size_t gn = ((size_t)e * FF + (size_t)(fc + kk + KB) + (tid >> 6)) * DD
                    + (tid & 63) * 16;
        __builtin_prefetch(w2 + gn, 0, 1);
      }
      __syncthreads();

      // A fragments from activation tile (rows 0-15 / 16-31)
      v8bf f0l = *(const v8bf*)&asb[lm][kk + lh * 16];
      v8bf f0h = *(const v8bf*)&asb[lm][kk + lh * 16 + 8];
      v16bf af0 = __builtin_shufflevector(f0l, f0h,
                                          0, 1, 2, 3, 4, 5, 6, 7,
                                          8, 9, 10, 11, 12, 13, 14, 15);
      v8bf f1l = *(const v8bf*)&asb[16 + lm][kk + lh * 16];
      v8bf f1h = *(const v8bf*)&asb[16 + lm][kk + lh * 16 + 8];
      v16bf af1 = __builtin_shufflevector(f1l, f1h,
                                          0, 1, 2, 3, 4, 5, 6, 7,
                                          8, 9, 10, 11, 12, 13, 14, 15);
#pragma unroll
      for (int tn = 0; tn < 8; ++tn) {
        v16bf bfr = *(const v16bf*)&w2sf[wave * 8 + tn][lane][0];
        oacc0[tn] = __builtin_amdgcn_wmma_f32_16x16x32_bf16(
            false, af0, false, bfr, (short)0, oacc0[tn], false, false);
        oacc1[tn] = __builtin_amdgcn_wmma_f32_16x16x32_bf16(
            false, af1, false, bfr, (short)0, oacc1[tn], false, false);
      }
      __syncthreads();
    }
  }

  // ------- weighted scatter-combine into out (f32 atomics) ---------------
#pragma unroll 1
  for (int tn = 0; tn < 8; ++tn) {
#pragma unroll
    for (int v = 0; v < 8; ++v) {
      int M = v + 8 * lh;
      int col = wave * 128 + tn * 16 + lm;
      int t0 = trow[M];
      if (t0 >= 0)
        atomicAdd(&out[(size_t)t0 * DD + col], oacc0[tn][v] * crow[M]);
      int t1 = trow[16 + M];
      if (t1 >= 0)
        atomicAdd(&out[(size_t)t1 * DD + col], oacc1[tn][v] * crow[16 + M]);
    }
  }
}

// ---------------------------------------------------------------------------
extern "C" void kernel_launch(void* const* d_in, const int* in_sizes, int n_in,
                              void* d_out, int out_size, void* d_ws, size_t ws_size,
                              hipStream_t stream) {
  (void)in_sizes; (void)n_in; (void)out_size; (void)ws_size;
  const float* x  = (const float*)d_in[0];  // [T, D]
  const float* gw = (const float*)d_in[1];  // [E, D]
  const float* w1 = (const float*)d_in[2];  // [E, D, F]
  const float* w2 = (const float*)d_in[3];  // [E, F, D]
  const float* w3 = (const float*)d_in[4];  // [E, D, F]

  float* out    = (float*)d_out;                       // [T, D]
  float* logits = out + (size_t)TT * DD;               // [T, E]

  char* ws = (char*)d_ws;
  int*   cnt     = (int*)ws;                                   // E ints
  int*   tokens  = (int*)(ws + 256);                           // E*T ints
  float* combine = (float*)(ws + 256 + (size_t)EE * TT * 4);   // T*E f32
  bf16*  xh      = (bf16*)(ws + 256 + (size_t)EE * TT * 4
                               + (size_t)TT * EE * 4);         // T*D bf16

  const int threads = 256;
  const int nElem   = TT * DD;

  moe_init_kernel<<<(nElem + threads - 1) / threads, threads, 0, stream>>>(
      out, combine, cnt);
  moe_cast_kernel<<<(nElem + threads - 1) / threads, threads, 0, stream>>>(
      x, xh);
  moe_router_kernel<<<TT / 8, threads, 0, stream>>>(
      x, gw, logits, combine, cnt, tokens);
  moe_expert_kernel<<<EE * (TT / BM), threads, 0, stream>>>(
      w1, w2, w3, xh, combine, cnt, tokens, out);
}
